// SAIGTransformer_77996606095740
// MI455X (gfx1250) — matrix operand order repaired and checked by reading, not compiled, not run on hardware
//
#include <hip/hip_runtime.h>
#include <math.h>

// ---------------- types ----------------
typedef __bf16 bf16;
typedef __attribute__((ext_vector_type(16))) __bf16 v16bf;
typedef __attribute__((ext_vector_type(8)))  float  v8f;
typedef __attribute__((ext_vector_type(4)))  int    i32x4;

#define HWPIX  65536
#define NB     4
#define CDIM   96
#define NPIX   (NB * HWPIX)      // 262144
#define QC     288

// ---------------- optional gfx1250 async-to-LDS path ----------------
#if defined(__has_builtin)
#if __has_builtin(__builtin_amdgcn_global_load_async_to_lds_b128) && \
    __has_builtin(__builtin_amdgcn_s_wait_asynccnt)
#define HAVE_ASYNC_LDS 1
#endif
#endif

#ifdef HAVE_ASYNC_LDS
typedef __attribute__((address_space(1))) i32x4* gv4p;   // global v4i32*
typedef __attribute__((address_space(3))) i32x4* lv4p;   // LDS    v4i32*
#endif

// ---------------- workspace layout (bytes) ----------------
static constexpr size_t SZ_XN   = (size_t)NPIX * 96 * 2;      // 50,331,648
static constexpr size_t SZ_QKV  = (size_t)NPIX * 288 * 2;     // 150,994,944
static constexpr size_t O_XN    = 0;
static constexpr size_t O_QKV0  = O_XN   + SZ_XN;             // reused later as g (NPIX*256*2 fits)
static constexpr size_t O_QKV1  = O_QKV0 + SZ_QKV;
static constexpr size_t O_SVPQ  = O_QKV1 + SZ_QKV;            // NPIX*12*2
static constexpr size_t O_X2    = O_SVPQ + (size_t)NPIX * 12 * 2;   // NPIX*96*4
static constexpr size_t O_SS    = O_X2   + (size_t)NPIX * 96 * 4;   // 1200 floats (pad 8192)
static constexpr size_t O_ATT   = O_SS   + 8192;              // 4*4*27*24 floats (pad 65536)
static constexpr size_t O_WB    = O_ATT  + 65536;
static constexpr size_t O_QKVB  = O_WB;                       // 288*96 bf16
static constexpr size_t O_W12B  = O_QKVB + 288 * 96 * 2;      // 512*96 bf16 (rows 255,511 zero)
static constexpr size_t O_FOB   = O_W12B + 512 * 96 * 2;      // 96*256 bf16 (col 255 zero)
static constexpr size_t O_MB    = O_FOB  + 96 * 256 * 2;      // 4*96*96 bf16
static constexpr size_t WB_ZERO = O_MB - O_WB;

// ---------------- cooperative weight staging into LDS ----------------
template <int BYTES>
__device__ __forceinline__ void stage_to_lds(const bf16* __restrict__ gsrc, bf16* lds) {
  const int n16 = BYTES / 16;
  int t = threadIdx.x;
#ifdef HAVE_ASYNC_LDS
  for (int i = t; i < n16; i += 256) {
    __builtin_amdgcn_global_load_async_to_lds_b128(
        (gv4p)((char*)gsrc + (size_t)i * 16),
        (lv4p)((char*)lds + (size_t)i * 16),
        0, 0);
  }
  __builtin_amdgcn_s_wait_asynccnt(0);
  __syncthreads();
#else
  const uint4* s = (const uint4*)gsrc;
  uint4* d = (uint4*)lds;
  for (int i = t; i < n16; i += 256) d[i] = s[i];
  __syncthreads();
#endif
}

// ---------------- WMMA helpers (CDNA5 16x16x32 bf16) ----------------
__device__ __forceinline__ v16bf load_a_tile(const bf16* __restrict__ A, int lda, int row0, int k0) {
  int lane = threadIdx.x & 31;
  const bf16* p = A + (size_t)(row0 + (lane & 15)) * lda + k0 + ((lane & 16) ? 8 : 0);
  v16bf a;
#pragma unroll
  for (int e = 0; e < 8; ++e) a[e] = p[e];
#pragma unroll
  for (int e = 0; e < 8; ++e) a[8 + e] = p[16 + e];
  return a;
}
__device__ __forceinline__ v16bf load_b_tile(const bf16* W, int ldw, int col0, int k0) {
  int lane = threadIdx.x & 31;
  const bf16* p = W + (size_t)(col0 + (lane & 15)) * ldw + k0 + ((lane & 16) ? 8 : 0);
  v16bf b;
#pragma unroll
  for (int e = 0; e < 8; ++e) b[e] = p[e];
#pragma unroll
  for (int e = 0; e < 8; ++e) b[8 + e] = p[16 + e];
  return b;
}
__device__ __forceinline__ v8f wmma_bf16(v16bf a, v16bf b, v8f c) {
  return __builtin_amdgcn_wmma_f32_16x16x32_bf16(false, a, false, b, (short)0, c, false, false);
}

// ---------------- weight prep: fp32 -> bf16, padded ----------------
__global__ void k_wprep(const float* __restrict__ qkv_w, const float* __restrict__ ffn1,
                        const float* __restrict__ ffn2, const float* __restrict__ ffn_out,
                        bf16* __restrict__ qkvb, bf16* __restrict__ w12b, bf16* __restrict__ fob) {
  int i = blockIdx.x * blockDim.x + threadIdx.x;
  if (i < 288 * 96) qkvb[i] = (bf16)qkv_w[i];
  if (i < 255 * 96) {
    w12b[i] = (bf16)ffn1[i];             // rows 0..254
    w12b[256 * 96 + i] = (bf16)ffn2[i];  // rows 256..510
  }
  if (i < 96 * 255) {
    int o = i / 255, h = i % 255;
    fob[o * 256 + h] = (bf16)ffn_out[i];
  }
}

// ---------------- channel LayerNorm -> bf16 pixel-major ----------------
__global__ void k_ln(const float* __restrict__ x, const float* __restrict__ w,
                     const float* __restrict__ bsh, bf16* __restrict__ out) {
  int p = blockIdx.x * blockDim.x + threadIdx.x;   // global pixel
  int b = p >> 16, ph = p & 65535;
  const float* xp = x + (size_t)b * 96 * HWPIX + ph;
  float v[96];
  float s = 0.f;
#pragma unroll
  for (int c = 0; c < 96; ++c) { v[c] = xp[(size_t)c * HWPIX]; s += v[c]; }
  float mu = s * (1.f / 96.f);
  float s2 = 0.f;
#pragma unroll
  for (int c = 0; c < 96; ++c) { float d = v[c] - mu; s2 += d * d; }
  float rstd = rsqrtf(s2 * (1.f / 96.f) + 1e-5f);
  bf16* op = out + (size_t)p * 96;
#pragma unroll
  for (int c = 0; c < 96; ++c)
    op[c] = (bf16)(((v[c] - mu) * rstd) * w[c] + bsh[c]);
}

// ---------------- svp 1x1 conv (12x3) ----------------
__global__ void k_svpq(const float* __restrict__ svp, const float* __restrict__ w,
                       bf16* __restrict__ out) {
  int p = blockIdx.x * blockDim.x + threadIdx.x;
  int b = p >> 16, ph = p & 65535;
  const float* sp = svp + (size_t)b * 3 * HWPIX + ph;
  float f0 = sp[0], f1 = sp[HWPIX], f2 = sp[2 * (size_t)HWPIX];
  bf16* op = out + (size_t)p * 12;
#pragma unroll
  for (int j = 0; j < 12; ++j)
    op[j] = (bf16)(w[j * 3] * f0 + w[j * 3 + 1] * f1 + w[j * 3 + 2] * f2);
}

// ---------------- GEMM: qkv = xn @ qkv_w^T (N=288, K=96); wave per m-tile ----------------
__global__ __launch_bounds__(256) void k_gemm_qkv(const bf16* __restrict__ xn,
                                                  const bf16* __restrict__ wb,
                                                  bf16* __restrict__ qkv0) {
  __shared__ bf16 sw[288 * 96];
  stage_to_lds<288 * 96 * 2>(wb, sw);
  int wave = (blockIdx.x * blockDim.x + threadIdx.x) >> 5;
  int row0 = wave * 16;
  v16bf a0 = load_a_tile(xn, 96, row0, 0);
  v16bf a1 = load_a_tile(xn, 96, row0, 32);
  v16bf a2 = load_a_tile(xn, 96, row0, 64);
  int lane = threadIdx.x & 31;
  int rb = row0 + ((lane & 16) ? 8 : 0);
  for (int n = 0; n < 18; ++n) {
    v8f acc = {};
    acc = wmma_bf16(a0, load_b_tile(sw, 96, n * 16, 0), acc);
    acc = wmma_bf16(a1, load_b_tile(sw, 96, n * 16, 32), acc);
    acc = wmma_bf16(a2, load_b_tile(sw, 96, n * 16, 64), acc);
    int col = n * 16 + (lane & 15);
#pragma unroll
    for (int r = 0; r < 8; ++r)
      qkv0[(size_t)(rb + r) * 288 + col] = (bf16)acc[r];
  }
}

// ---------------- depthwise 3x3 'SAME' on (pixel,288) bf16 ----------------
__global__ void k_dw(const bf16* __restrict__ qkv0, const float* __restrict__ wdw,
                     bf16* __restrict__ qkv1) {
  size_t t = (size_t)blockIdx.x * blockDim.x + threadIdx.x;
  int c = (int)(t % 288);
  size_t pix = t / 288;
  int b = (int)(pix >> 16), ph = (int)(pix & 65535);
  int y = ph >> 8, xc = ph & 255;
  const float* wp = wdw + c * 9;
  float acc = 0.f;
#pragma unroll
  for (int dy = -1; dy <= 1; ++dy) {
    int yy = y + dy; if (yy < 0 || yy > 255) continue;
#pragma unroll
    for (int dx = -1; dx <= 1; ++dx) {
      int xx = xc + dx; if (xx < 0 || xx > 255) continue;
      size_t q = ((size_t)b << 16) + (yy << 8) + xx;
      acc += wp[(dy + 1) * 3 + (dx + 1)] * (float)qkv0[q * 288 + c];
    }
  }
  qkv1[pix * 288 + c] = (bf16)acc;
}

// ---------------- per-channel sum of squares (for l2 norms) ----------------
__global__ void k_sumsq(const bf16* __restrict__ qkv1, const bf16* __restrict__ svpq,
                        float* __restrict__ ss) {
  int chan = blockIdx.x;               // 0 .. NB*300-1
  int b = chan / 300, c = chan % 300;
  float acc = 0.f;
  if (c < 288) {
    const bf16* p = qkv1 + (size_t)b * HWPIX * 288 + c;
    for (int i = threadIdx.x; i < HWPIX; i += 256) { float v = (float)p[(size_t)i * 288]; acc += v * v; }
  } else {
    const bf16* p = svpq + (size_t)b * HWPIX * 12 + (c - 288);
    for (int i = threadIdx.x; i < HWPIX; i += 256) { float v = (float)p[(size_t)i * 12]; acc += v * v; }
  }
  __shared__ float red[256];
  red[threadIdx.x] = acc; __syncthreads();
  for (int s = 128; s > 0; s >>= 1) { if (threadIdx.x < (unsigned)s) red[threadIdx.x] += red[threadIdx.x + s]; __syncthreads(); }
  if (threadIdx.x == 0) ss[chan] = red[0];
}

// ---------------- Gram: G[b,h] = [q;svp](27 x hw) @ k^T(hw x 24) ----------------
__global__ __launch_bounds__(256) void k_gram(const bf16* __restrict__ qkv1,
                                              const bf16* __restrict__ svpq,
                                              float* __restrict__ att) {
  int bh = blockIdx.x, b = bh >> 2, head = bh & 3;
  int pbase0 = blockIdx.y * 2048;
  __shared__ float sm[51 * 128];       // rows 0..23 q, 24..26 svp, 27..50 k
  int t = threadIdx.x;
  int p0 = t, p1 = t + 256, p2 = t + 512;   // (i,d) pairs, i=pair/24, d=pair%24
  float a0 = 0.f, a1 = 0.f, a2 = 0.f;
  for (int chunk = 0; chunk < 16; ++chunk) {
    int pbase = pbase0 + chunk * 128;
    __syncthreads();
    for (int idx = t; idx < 51 * 128; idx += 256) {
      int r = idx >> 7, pp = idx & 127;
      size_t pg = (size_t)b * HWPIX + pbase + pp;
      float v;
      if (r < 24)      v = (float)qkv1[pg * 288 + head * 24 + r];
      else if (r < 27) v = (float)svpq[pg * 12 + head * 3 + (r - 24)];
      else             v = (float)qkv1[pg * 288 + 96 + head * 24 + (r - 27)];
      sm[idx] = v;
    }
    __syncthreads();
    {
      const float* q = &sm[(p0 / 24) * 128];
      const float* kk = &sm[(27 + p0 % 24) * 128];
      float s = 0.f;
      for (int pp = 0; pp < 128; ++pp) s += q[pp] * kk[pp];
      a0 += s;
    }
    if (p1 < 648) {
      const float* q = &sm[(p1 / 24) * 128];
      const float* kk = &sm[(27 + p1 % 24) * 128];
      float s = 0.f;
      for (int pp = 0; pp < 128; ++pp) s += q[pp] * kk[pp];
      a1 += s;
    }
    if (p2 < 648) {
      const float* q = &sm[(p2 / 24) * 128];
      const float* kk = &sm[(27 + p2 % 24) * 128];
      float s = 0.f;
      for (int pp = 0; pp < 128; ++pp) s += q[pp] * kk[pp];
      a2 += s;
    }
  }
  float* gp = att + (size_t)bh * 648;
  atomicAdd(&gp[p0], a0);
  if (p1 < 648) atomicAdd(&gp[p1], a1);
  if (p2 < 648) atomicAdd(&gp[p2], a2);
}

// ---------------- norms + softmax + fold proj: M_b (96x96) ----------------
__global__ void k_smax_mb(const float* __restrict__ att, const float* __restrict__ ss,
                          const float* __restrict__ temp, const float* __restrict__ proj,
                          bf16* __restrict__ mb) {
  int b = blockIdx.x;
  __shared__ float attn[108][24];
  int t = threadIdx.x;                   // 128 threads
  if (t < 108) {
    int head = t / 27, i = t % 27;
    float nq = (i < 24) ? sqrtf(ss[b * 300 + head * 24 + i])
                        : sqrtf(ss[b * 300 + 288 + head * 3 + (i - 24)]);
    nq = fmaxf(nq, 1e-12f);
    float tm = temp[head];
    float row[24]; float mx = -1e30f;
    for (int d = 0; d < 24; ++d) {
      float nk = fmaxf(sqrtf(ss[b * 300 + 96 + head * 24 + d]), 1e-12f);
      float v = att[((size_t)b * 4 + head) * 648 + i * 24 + d] * tm / (nq * nk);
      row[d] = v; mx = fmaxf(mx, v);
    }
    float sum = 0.f;
    for (int d = 0; d < 24; ++d) { row[d] = __expf(row[d] - mx); sum += row[d]; }
    float inv = 1.f / sum;
    for (int d = 0; d < 24; ++d) attn[t][d] = row[d] * inv;
  }
  __syncthreads();
  if (t < 96) {
    for (int head = 0; head < 4; ++head)
      for (int d = 0; d < 24; ++d) {
        float s = 0.f;
        for (int i = 0; i < 27; ++i)
          s += proj[t * 108 + head * 27 + i] * attn[head * 27 + i][d];
        mb[((size_t)b * 96 + t) * 96 + head * 24 + d] = (bf16)s;
      }
  }
}

// ---------------- GEMM: x2 = x + M_b @ v (N=96, K=96); wave per m-tile ----------------
__global__ __launch_bounds__(256) void k_gemm_mb(const bf16* __restrict__ qkv1,
                                                 const bf16* __restrict__ mb,
                                                 const float* __restrict__ x,
                                                 float* __restrict__ x2) {
  __shared__ bf16 sw[96 * 96];
  int bblk = (blockIdx.x * 128) >> 16;               // batch of this block
  stage_to_lds<96 * 96 * 2>(mb + (size_t)bblk * 96 * 96, sw);
  int wave = (blockIdx.x * blockDim.x + threadIdx.x) >> 5;
  int row0 = wave * 16;
  int b = row0 >> 16;
  const bf16* A = qkv1 + (size_t)row0 * 288 + 192;   // v block
  v16bf a0 = load_a_tile(A, 288, 0, 0);
  v16bf a1 = load_a_tile(A, 288, 0, 32);
  v16bf a2 = load_a_tile(A, 288, 0, 64);
  int lane = threadIdx.x & 31;
  int pp = (row0 & 65535) + ((lane & 16) ? 8 : 0);
  for (int n = 0; n < 6; ++n) {
    v8f acc = {};
    acc = wmma_bf16(a0, load_b_tile(sw, 96, n * 16, 0), acc);
    acc = wmma_bf16(a1, load_b_tile(sw, 96, n * 16, 32), acc);
    acc = wmma_bf16(a2, load_b_tile(sw, 96, n * 16, 64), acc);
    int ch = n * 16 + (lane & 15);
    size_t base = ((size_t)b * 96 + ch) * HWPIX + pp;
#pragma unroll
    for (int r = 0; r < 8; ++r) x2[base + r] = x[base + r] + acc[r];
  }
}

// ---------------- GEMM ffn1 + fused dual gating -> g (pixel x 256) ----------------
__global__ __launch_bounds__(256) void k_gemm_ffn1(const bf16* __restrict__ xn2,
                                                   const bf16* __restrict__ w12,
                                                   bf16* __restrict__ g) {
  __shared__ bf16 sw[256 * 96];                      // p1 half (rows 0..255)
  stage_to_lds<256 * 96 * 2>(w12, sw);
  const bf16* w2 = w12 + 256 * 96;                   // p2 half stays in L2
  int wave = (blockIdx.x * blockDim.x + threadIdx.x) >> 5;
  int row0 = wave * 16;
  v16bf a0 = load_a_tile(xn2, 96, row0, 0);
  v16bf a1 = load_a_tile(xn2, 96, row0, 32);
  v16bf a2 = load_a_tile(xn2, 96, row0, 64);
  int lane = threadIdx.x & 31;
  int rb = row0 + ((lane & 16) ? 8 : 0);
  for (int j = 0; j < 16; ++j) {
    v8f acc1 = {}, acc2 = {};
    acc1 = wmma_bf16(a0, load_b_tile(sw, 96, j * 16, 0), acc1);
    acc1 = wmma_bf16(a1, load_b_tile(sw, 96, j * 16, 32), acc1);
    acc1 = wmma_bf16(a2, load_b_tile(sw, 96, j * 16, 64), acc1);
    acc2 = wmma_bf16(a0, load_b_tile(w2, 96, j * 16, 0), acc2);
    acc2 = wmma_bf16(a1, load_b_tile(w2, 96, j * 16, 32), acc2);
    acc2 = wmma_bf16(a2, load_b_tile(w2, 96, j * 16, 64), acc2);
    int col = j * 16 + (lane & 15);
#pragma unroll
    for (int r = 0; r < 8; ++r) {
      float v1 = acc1[r], v2 = acc2[r];
      float sig = 1.f / (1.f + __expf(-v2));
      float gel = 0.5f * v1 * (1.f + erff(v1 * 0.70710678118f));
      g[(size_t)(rb + r) * 256 + col] = (bf16)(v1 * sig + v2 * gel);
    }
  }
}

// ---------------- GEMM ffn_out: out = x2 + g @ fob^T (N=96, K=256) ----------------
__global__ __launch_bounds__(256) void k_gemm_ffn2(const bf16* __restrict__ g,
                                                   const bf16* __restrict__ fob,
                                                   const float* __restrict__ x2,
                                                   float* __restrict__ out) {
  __shared__ bf16 sw[96 * 256];
  stage_to_lds<96 * 256 * 2>(fob, sw);
  int wave = (blockIdx.x * blockDim.x + threadIdx.x) >> 5;
  int row0 = wave * 16;
  int b = row0 >> 16;
  v16bf a[8];
#pragma unroll
  for (int kk = 0; kk < 8; ++kk) a[kk] = load_a_tile(g, 256, row0, kk * 32);
  int lane = threadIdx.x & 31;
  int pp = (row0 & 65535) + ((lane & 16) ? 8 : 0);
  for (int n = 0; n < 6; ++n) {
    v8f acc = {};
#pragma unroll
    for (int kk = 0; kk < 8; ++kk)
      acc = wmma_bf16(a[kk], load_b_tile(sw, 256, n * 16, kk * 32), acc);
    int ch = n * 16 + (lane & 15);
    size_t base = ((size_t)b * 96 + ch) * HWPIX + pp;
#pragma unroll
    for (int r = 0; r < 8; ++r) out[base + r] = x2[base + r] + acc[r];
  }
}

// ---------------- launch ----------------
extern "C" void kernel_launch(void* const* d_in, const int* in_sizes, int n_in,
                              void* d_out, int out_size, void* d_ws, size_t ws_size,
                              hipStream_t stream) {
  const float* x        = (const float*)d_in[0];
  const float* svp_fea  = (const float*)d_in[1];
  const float* norm1_w  = (const float*)d_in[2];
  const float* norm1_b  = (const float*)d_in[3];
  const float* qkv_w    = (const float*)d_in[4];
  const float* qkv_dw_w = (const float*)d_in[5];
  const float* svp_w    = (const float*)d_in[6];
  const float* temper   = (const float*)d_in[7];
  const float* proj_w   = (const float*)d_in[8];
  const float* norm2_w  = (const float*)d_in[9];
  const float* norm2_b  = (const float*)d_in[10];
  const float* ffn1_w   = (const float*)d_in[11];
  const float* ffn2_w   = (const float*)d_in[12];
  const float* ffnout_w = (const float*)d_in[13];

  char* ws = (char*)d_ws;
  bf16*  xn    = (bf16*)(ws + O_XN);
  bf16*  qkv0  = (bf16*)(ws + O_QKV0);   // pre-dw qkv; later reused as g
  bf16*  qkv1  = (bf16*)(ws + O_QKV1);
  bf16*  svpq  = (bf16*)(ws + O_SVPQ);
  float* x2    = (float*)(ws + O_X2);
  float* ss    = (float*)(ws + O_SS);
  float* att   = (float*)(ws + O_ATT);
  bf16*  qkvb  = (bf16*)(ws + O_QKVB);
  bf16*  w12b  = (bf16*)(ws + O_W12B);
  bf16*  fob   = (bf16*)(ws + O_FOB);
  bf16*  mb    = (bf16*)(ws + O_MB);
  bf16*  g     = (bf16*)(ws + O_QKV0);   // reuse
  float* out   = (float*)d_out;

  // zero atomic target + padded weight region (graph-capturable memsets)
  (void)hipMemsetAsync(ws + O_ATT, 0, 65536, stream);
  (void)hipMemsetAsync(ws + O_WB, 0, WB_ZERO, stream);

  // weight conversion
  k_wprep<<<108, 256, 0, stream>>>(qkv_w, ffn1_w, ffn2_w, ffnout_w, qkvb, w12b, fob);

  // attention branch
  k_ln<<<NPIX / 256, 256, 0, stream>>>(x, norm1_w, norm1_b, xn);
  k_svpq<<<NPIX / 256, 256, 0, stream>>>(svp_fea, svp_w, svpq);
  k_gemm_qkv<<<NPIX / 16 / 8, 256, 0, stream>>>(xn, qkvb, qkv0);          // 2048 blocks
  k_dw<<<NPIX * (QC / 256), 256, 0, stream>>>(qkv0, qkv_dw_w, qkv1);      // 262144 blocks
  k_sumsq<<<NB * 300, 256, 0, stream>>>(qkv1, svpq, ss);
  k_gram<<<dim3(NB * 4, 32), 256, 0, stream>>>(qkv1, svpq, att);
  k_smax_mb<<<NB, 128, 0, stream>>>(att, ss, temper, proj_w, mb);
  k_gemm_mb<<<NPIX / 16 / 8, 256, 0, stream>>>(qkv1, mb, x, x2);

  // FFN branch
  k_ln<<<NPIX / 256, 256, 0, stream>>>(x2, norm2_w, norm2_b, xn);         // xn reused as xn2
  k_gemm_ffn1<<<NPIX / 16 / 8, 256, 0, stream>>>(xn, w12b, g);
  k_gemm_ffn2<<<NPIX / 16 / 8, 256, 0, stream>>>(g, fob, x2, out);

  (void)in_sizes; (void)n_in; (void)out_size; (void)ws_size;
}
